// MultiHeadAttention_84748294685498
// MI455X (gfx1250) — matrix operand order repaired and checked by reading, compiled, not run on hardware
//
#include <hip/hip_runtime.h>
#include <hip/hip_bf16.h>
#include <math.h>

typedef __bf16 bf16_t;
typedef __bf16 v16bf __attribute__((ext_vector_type(16)));
typedef __bf16 v8bf  __attribute__((ext_vector_type(8)));
typedef float  v8f   __attribute__((ext_vector_type(8)));

#define D_MODEL 1024
#define SEQ     2048
#define BATCH   2
#define HEADS   16
#define DK      64
#define ROWS    (BATCH * SEQ)   /* 4096 */

#define QSTRIP  16               /* queries per fused-attention workgroup */
#define LDS_LD  2052             /* padded row stride (floats): 2052%64==4 -> conflict-free
                                    column reads across 16 rows; 2052*4 bytes % 16 == 0 */

// ---------------------------------------------------------------------------
// WMMA helpers (CDNA5 gfx1250, wave32). Layouts per CDNA5 ISA §7.12.2
// ---------------------------------------------------------------------------
__device__ __forceinline__ v8f wmma_bf16(v16bf a, v16bf b, v8f c) {
  return __builtin_amdgcn_wmma_f32_16x16x32_bf16(
      false, a, false, b, (short)0, c, false, false);
}

// A fragment: 16x32 (MxK) bf16 from row-major global memory.
__device__ __forceinline__ v16bf load_frag_a(const bf16_t* A, int lda, int m0, int k0) {
  const int lane = threadIdx.x & 31;
  const int m  = lane & 15;
  const int kh = lane >> 4;
  const bf16_t* p = A + (size_t)(m0 + m) * lda + (k0 + kh * 8);
  union { v16bf v; struct { v8bf lo, hi; } s; } u;
  u.s.lo = *(const v8bf*)(p);        // K = k0 + kh*8 + 0..7
  u.s.hi = *(const v8bf*)(p + 16);   // K = k0 + 16 + kh*8 + 0..7
  return u.v;
}

// A fragment sourced from the f32 score strip in LDS (row = lane&15), cvt->bf16.
__device__ __forceinline__ v16bf load_frag_a_lds(const float* sS, int k0) {
  const int lane = threadIdx.x & 31;
  const int m  = lane & 15;
  const int kh = lane >> 4;
  const float* p = sS + m * LDS_LD + (k0 + kh * 8);
  float4 lo0 = *(const float4*)(p);
  float4 lo1 = *(const float4*)(p + 4);
  float4 hi0 = *(const float4*)(p + 16);
  float4 hi1 = *(const float4*)(p + 20);
  v16bf r;
  r[0]  = (bf16_t)lo0.x; r[1]  = (bf16_t)lo0.y; r[2]  = (bf16_t)lo0.z; r[3]  = (bf16_t)lo0.w;
  r[4]  = (bf16_t)lo1.x; r[5]  = (bf16_t)lo1.y; r[6]  = (bf16_t)lo1.z; r[7]  = (bf16_t)lo1.w;
  r[8]  = (bf16_t)hi0.x; r[9]  = (bf16_t)hi0.y; r[10] = (bf16_t)hi0.z; r[11] = (bf16_t)hi0.w;
  r[12] = (bf16_t)hi1.x; r[13] = (bf16_t)hi1.y; r[14] = (bf16_t)hi1.z; r[15] = (bf16_t)hi1.w;
  return r;
}

// B fragment: 32x16 (KxN) bf16 from "Bt" (row n = B column n, contiguous in K).
__device__ __forceinline__ v16bf load_frag_b(const bf16_t* Bt, int ldb, int n0, int k0) {
  const int lane = threadIdx.x & 31;
  const int n  = lane & 15;
  const int kb = (lane >> 4) * 8;
  const bf16_t* p = Bt + (size_t)(n0 + n) * ldb + (k0 + kb);
  union { v16bf v; struct { v8bf lo, hi; } s; } u;
  u.s.lo = *(const v8bf*)(p);
  u.s.hi = *(const v8bf*)(p + 16);
  return u.v;
}

// ---------------------------------------------------------------------------
// Elementwise f32 -> bf16
// ---------------------------------------------------------------------------
__global__ void cvt_f32_bf16(const float* __restrict__ in, bf16_t* __restrict__ out, int n) {
  int i = blockIdx.x * blockDim.x + threadIdx.x;
  const int stride = gridDim.x * blockDim.x;
  for (; i < n; i += stride) out[i] = (bf16_t)in[i];
}

// ---------------------------------------------------------------------------
// Projection GEMM: C[m,n] = sum_k A_bf16[m,k] * W_bf16[n,k]
// mode 0: bf16 row-major [ROWS,1024]; mode 1: Vt[b][h][d][s]
// ---------------------------------------------------------------------------
__global__ void gemm_qkv(const bf16_t* __restrict__ A, const bf16_t* __restrict__ Bt,
                         bf16_t* __restrict__ C, int mode) {
  const int n0 = blockIdx.x * 32;
  const int m0 = blockIdx.y * 32;
  v8f acc[2][2] = {};
  for (int k0 = 0; k0 < D_MODEL; k0 += 32) {
    __builtin_prefetch(A + (size_t)m0 * D_MODEL + k0 + 64, 0, 1);
    v16bf a0 = load_frag_a(A, D_MODEL, m0,      k0);
    v16bf a1 = load_frag_a(A, D_MODEL, m0 + 16, k0);
    v16bf b0 = load_frag_b(Bt, D_MODEL, n0,      k0);
    v16bf b1 = load_frag_b(Bt, D_MODEL, n0 + 16, k0);
    acc[0][0] = wmma_bf16(a0, b0, acc[0][0]);
    acc[0][1] = wmma_bf16(a0, b1, acc[0][1]);
    acc[1][0] = wmma_bf16(a1, b0, acc[1][0]);
    acc[1][1] = wmma_bf16(a1, b1, acc[1][1]);
  }
  const int lane = threadIdx.x & 31;
  const int nn   = lane & 15;
  const int mtop = (lane >> 4) * 8;
#pragma unroll
  for (int mi = 0; mi < 2; ++mi)
#pragma unroll
    for (int nt = 0; nt < 2; ++nt)
#pragma unroll
      for (int j = 0; j < 8; ++j) {
        const int m = m0 + mi * 16 + mtop + j;
        const int n = n0 + nt * 16 + nn;
        const float val = acc[mi][nt][j];
        if (mode == 0) {
          C[(size_t)m * D_MODEL + n] = (bf16_t)val;
        } else {  // Vt[b][h][d][s]
          const int b = m >> 11, s = m & (SEQ - 1);
          const int h = n >> 6,  d = n & (DK - 1);
          C[((((size_t)b * HEADS + h) * DK + d) << 11) + s] = (bf16_t)val;
        }
      }
}

// ---------------------------------------------------------------------------
// Fused attention core: one 256-thread (8-wave) workgroup per (b,h, 16-query
// strip). Scores (WMMA) -> LDS (128 KB) -> softmax in LDS -> single attn
// write to d_out -> context GEMM with A streamed from LDS.
// Exploits CDNA5's 320 KB-per-WGP LDS to cut attn HBM traffic ~4x.
// ---------------------------------------------------------------------------
__global__ void __launch_bounds__(256)
attn_fused(const bf16_t* __restrict__ Qb, const bf16_t* __restrict__ Kb,
           const bf16_t* __restrict__ Vt, float* __restrict__ attn_out,
           bf16_t* __restrict__ ctx) {
  __shared__ float sS[QSTRIP * LDS_LD];        // 16 x 2048 scores (padded) ~128.25 KB
  __shared__ float sRed[2][QSTRIP][16];        // row max / row sum partials
  __shared__ float sAcc[4][QSTRIP][16];        // context K-half combine

  const int bh = blockIdx.y;                   // 0..31
  const int b = bh >> 4, h = bh & 15;
  const int m0 = blockIdx.x * QSTRIP;          // global query base of strip
  const int tid  = threadIdx.x;
  const int wave = tid >> 5;                   // 0..7
  const int lane = tid & 31;
  const int nn   = lane & 15;
  const int mtop = (lane >> 4) * 8;

  const bf16_t* Aq = Qb + (size_t)b * SEQ * D_MODEL + h * DK;
  const bf16_t* Bk = Kb + (size_t)b * SEQ * D_MODEL + h * DK;

  // ---- Phase 1: score strip into LDS (wave w owns keys [w*256, w*256+256)) --
  {
    v16bf a0 = load_frag_a(Aq, D_MODEL, m0, 0);   // reused for all key tiles
    v16bf a1 = load_frag_a(Aq, D_MODEL, m0, 32);
    for (int n0 = wave * 256; n0 < wave * 256 + 256; n0 += 32) {
      v8f acc0 = {}, acc1 = {};
      v16bf b00 = load_frag_b(Bk, D_MODEL, n0,      0);
      v16bf b01 = load_frag_b(Bk, D_MODEL, n0,      32);
      v16bf b10 = load_frag_b(Bk, D_MODEL, n0 + 16, 0);
      v16bf b11 = load_frag_b(Bk, D_MODEL, n0 + 16, 32);
      acc0 = wmma_bf16(a0, b00, acc0);
      acc0 = wmma_bf16(a1, b01, acc0);
      acc1 = wmma_bf16(a0, b10, acc1);
      acc1 = wmma_bf16(a1, b11, acc1);
#pragma unroll
      for (int j = 0; j < 8; ++j) {
        sS[(mtop + j) * LDS_LD + n0 + nn]      = acc0[j] * 0.125f;
        sS[(mtop + j) * LDS_LD + n0 + 16 + nn] = acc1[j] * 0.125f;
      }
    }
  }
  __syncthreads();

  // ---- Phase 2: softmax over the 16 LDS rows (16 threads per row) ----------
  const int r   = tid >> 4;                    // row 0..15
  const int sub = tid & 15;                    // sub-lane within row
  float* rowp = sS + r * LDS_LD;
  float mx = -3.402823466e+38f;
  for (int c = sub; c < SEQ; c += 16) mx = fmaxf(mx, rowp[c]);
  sRed[0][r][sub] = mx;
  __syncthreads();
  mx = sRed[0][r][0];
#pragma unroll
  for (int k = 1; k < 16; ++k) mx = fmaxf(mx, sRed[0][r][k]);
  float sum = 0.f;
  for (int c = sub; c < SEQ; c += 16) {
    const float e = __expf(rowp[c] - mx);
    rowp[c] = e;
    sum += e;
  }
  sRed[1][r][sub] = sum;
  __syncthreads();
  sum = 0.f;
#pragma unroll
  for (int k = 0; k < 16; ++k) sum += sRed[1][r][k];
  const float inv = 1.0f / sum;
  for (int c = sub; c < SEQ; c += 16) rowp[c] *= inv;
  __syncthreads();

  // ---- Phase 3: single coalesced attn write to d_out -----------------------
  {
    float* gattn = attn_out + (size_t)bh * SEQ * SEQ + (size_t)(m0 + r) * SEQ;
    for (int c4 = sub * 4; c4 < SEQ; c4 += 64) {
      float4 v = *(const float4*)(rowp + c4);
      *(float4*)(gattn + c4) = v;
    }
  }

  // ---- Phase 4: context = attn(LDS) @ V; 4 d-tiles x 2 K-halves ------------
  {
    const bf16_t* Bv = Vt + (size_t)bh * DK * SEQ;   // row d, ld = SEQ
    const int d0  = (wave & 3) * 16;
    const int kh0 = (wave >> 2) * 1024;
    v8f cacc = {};
    for (int k0 = kh0; k0 < kh0 + 1024; k0 += 32) {
      __builtin_prefetch(Bv + (size_t)d0 * SEQ + k0 + 64, 0, 1);
      v16bf af = load_frag_a_lds(sS, k0);
      v16bf bf_ = load_frag_b(Bv, SEQ, d0, k0);
      cacc = wmma_bf16(af, bf_, cacc);
    }
    if (wave >= 4) {
#pragma unroll
      for (int j = 0; j < 8; ++j) sAcc[wave & 3][mtop + j][nn] = cacc[j];
    }
    __syncthreads();
    if (wave < 4) {
#pragma unroll
      for (int j = 0; j < 8; ++j) {
        const float v = cacc[j] + sAcc[wave][mtop + j][nn];
        ctx[((size_t)b * SEQ + m0 + mtop + j) * D_MODEL + h * DK + d0 + nn] = (bf16_t)v;
      }
    }
  }
}

// ---------------------------------------------------------------------------
// Output projection: proj[m,n] = sum_k ctx[m,k] * Wo[n,k] + b_o[n]   (f32 out)
// ---------------------------------------------------------------------------
__global__ void gemm_oproj(const bf16_t* __restrict__ A, const bf16_t* __restrict__ Bt,
                           const float* __restrict__ bias, float* __restrict__ C) {
  const int n0 = blockIdx.x * 32;
  const int m0 = blockIdx.y * 32;
  v8f acc[2][2] = {};
  for (int k0 = 0; k0 < D_MODEL; k0 += 32) {
    __builtin_prefetch(A + (size_t)m0 * D_MODEL + k0 + 64, 0, 1);
    v16bf a0 = load_frag_a(A, D_MODEL, m0,      k0);
    v16bf a1 = load_frag_a(A, D_MODEL, m0 + 16, k0);
    v16bf b0 = load_frag_b(Bt, D_MODEL, n0,      k0);
    v16bf b1 = load_frag_b(Bt, D_MODEL, n0 + 16, k0);
    acc[0][0] = wmma_bf16(a0, b0, acc[0][0]);
    acc[0][1] = wmma_bf16(a0, b1, acc[0][1]);
    acc[1][0] = wmma_bf16(a1, b0, acc[1][0]);
    acc[1][1] = wmma_bf16(a1, b1, acc[1][1]);
  }
  const int lane = threadIdx.x & 31;
  const int nn   = lane & 15;
  const int mtop = (lane >> 4) * 8;
#pragma unroll
  for (int mi = 0; mi < 2; ++mi)
#pragma unroll
    for (int nt = 0; nt < 2; ++nt) {
      const int n = n0 + nt * 16 + nn;
      const float bo = bias[n];
#pragma unroll
      for (int j = 0; j < 8; ++j)
        C[(size_t)(m0 + mi * 16 + mtop + j) * D_MODEL + n] = acc[mi][nt][j] + bo;
    }
}

// ---------------------------------------------------------------------------
// Residual + LayerNorm: y = LN(proj + x) * gamma + beta ; one block per row.
// ---------------------------------------------------------------------------
__global__ void residual_ln(const float* __restrict__ proj, const float* __restrict__ x,
                            const float* __restrict__ gamma, const float* __restrict__ beta,
                            float* __restrict__ y) {
  __shared__ float red[256];
  const size_t row = blockIdx.x;
  const int tid = threadIdx.x;
  const float* pp = proj + row * D_MODEL;
  const float* px = x + row * D_MODEL;
  float v[4];
  float s = 0.f;
#pragma unroll
  for (int i = 0; i < 4; ++i) { v[i] = pp[tid + i * 256] + px[tid + i * 256]; s += v[i]; }
  red[tid] = s; __syncthreads();
  for (int o = 128; o > 0; o >>= 1) {
    if (tid < o) red[tid] += red[tid + o];
    __syncthreads();
  }
  const float mu = red[0] * (1.0f / D_MODEL);
  __syncthreads();
  s = 0.f;
#pragma unroll
  for (int i = 0; i < 4; ++i) { const float d = v[i] - mu; s += d * d; }
  red[tid] = s; __syncthreads();
  for (int o = 128; o > 0; o >>= 1) {
    if (tid < o) red[tid] += red[tid + o];
    __syncthreads();
  }
  const float rs = rsqrtf(red[0] * (1.0f / D_MODEL) + 1e-5f);
#pragma unroll
  for (int i = 0; i < 4; ++i) {
    const int c = tid + i * 256;
    y[row * D_MODEL + c] = (v[i] - mu) * rs * gamma[c] + beta[c];
  }
}

// ---------------------------------------------------------------------------
// Host launcher
// ---------------------------------------------------------------------------
extern "C" void kernel_launch(void* const* d_in, const int* in_sizes, int n_in,
                              void* d_out, int out_size, void* d_ws, size_t ws_size,
                              hipStream_t stream) {
  (void)in_sizes; (void)n_in; (void)out_size; (void)ws_size;
  const float* x     = (const float*)d_in[0];
  const float* Wq    = (const float*)d_in[1];
  const float* Wk    = (const float*)d_in[2];
  const float* Wv    = (const float*)d_in[3];
  const float* Wo    = (const float*)d_in[4];
  const float* b_o   = (const float*)d_in[5];
  const float* gamma = (const float*)d_in[6];
  const float* beta  = (const float*)d_in[7];

  float* y_out = (float*)d_out;                               // [2,2048,1024]
  float* attn  = y_out + (size_t)ROWS * D_MODEL;              // [2,16,2048,2048]

  // Workspace carve-up (~64 MB)
  bf16_t* xb  = (bf16_t*)d_ws;                                // 4096*1024 bf16
  bf16_t* wqb = xb  + (size_t)ROWS * D_MODEL;
  bf16_t* wkb = wqb + (size_t)D_MODEL * D_MODEL;
  bf16_t* wvb = wkb + (size_t)D_MODEL * D_MODEL;
  bf16_t* wob = wvb + (size_t)D_MODEL * D_MODEL;
  bf16_t* Qb  = wob + (size_t)D_MODEL * D_MODEL;              // [B,S,1024]
  bf16_t* Kb  = Qb  + (size_t)ROWS * D_MODEL;                 // [B,S,1024]
  bf16_t* Vt  = Kb  + (size_t)ROWS * D_MODEL;                 // [B,H,64,S]
  bf16_t* ctx = Vt  + (size_t)ROWS * D_MODEL;                 // [B,S,1024]
  float*  prj = (float*)(ctx + (size_t)ROWS * D_MODEL);       // [B,S,1024] f32

  // 1. bf16 conversions
  cvt_f32_bf16<<<1024, 256, 0, stream>>>(x,  xb,  ROWS * D_MODEL);
  cvt_f32_bf16<<<512,  256, 0, stream>>>(Wq, wqb, D_MODEL * D_MODEL);
  cvt_f32_bf16<<<512,  256, 0, stream>>>(Wk, wkb, D_MODEL * D_MODEL);
  cvt_f32_bf16<<<512,  256, 0, stream>>>(Wv, wvb, D_MODEL * D_MODEL);
  cvt_f32_bf16<<<512,  256, 0, stream>>>(Wo, wob, D_MODEL * D_MODEL);

  // 2. Q/K/V projections (WMMA); V stored transposed per head
  dim3 gProj(D_MODEL / 32, ROWS / 32);                        // (32,128)
  gemm_qkv<<<gProj, 32, 0, stream>>>(xb, wqb, Qb, 0);
  gemm_qkv<<<gProj, 32, 0, stream>>>(xb, wkb, Kb, 0);
  gemm_qkv<<<gProj, 32, 0, stream>>>(xb, wvb, Vt, 1);

  // 3. Fused scores + softmax + attn-write + context (LDS-resident strip)
  dim3 gA(SEQ / QSTRIP, BATCH * HEADS);                       // (128, 32)
  attn_fused<<<gA, 256, 0, stream>>>(Qb, Kb, Vt, attn, ctx);

  // 4. Output projection + bias (WMMA, f32 out)
  gemm_oproj<<<gProj, 32, 0, stream>>>(ctx, wob, b_o, prj);

  // 5. Residual + LayerNorm into d_out y slice
  residual_ln<<<ROWS, 256, 0, stream>>>(prj, x, gamma, beta, y_out);
}
